// AdMSoftmaxLoss_1099511628282
// MI455X (gfx1250) — matrix operand order repaired and checked by reading, not compiled, not run on hardware
//
#include <hip/hip_runtime.h>
#include <stdint.h>

#define NROWS 4096
#define DIM   512
#define NCLS  100000
#define SSCALE 30.0f
#define MARGINC 0.4f

typedef __attribute__((ext_vector_type(16))) __bf16 bf16x16;
typedef __attribute__((ext_vector_type(8)))  float  f32x8;

struct alignas(16) U128 { uint32_t x, y, z, w; };
union FragU { U128 u[2]; bf16x16 v; };

// round-to-nearest-even f32 -> bf16 (on raw bits)
__device__ inline uint32_t bf_rne(uint32_t u) {
  return (u + 0x7FFFu + ((u >> 16) & 1u)) >> 16;
}
__device__ inline uint32_t f2bf(float f) { return bf_rne(__float_as_uint(f)); }
__device__ inline float bflo(uint32_t u) { return __uint_as_float(u << 16); }
__device__ inline float bfhi(uint32_t u) { return __uint_as_float(u & 0xFFFF0000u); }

// ---------------------------------------------------------------- zero rowsum
__global__ void zero_kernel(float* __restrict__ rowsum) {
  int i = blockIdx.x * 256 + threadIdx.x;
  rowsum[i] = 0.0f;
}

// ------------------------------------------------- xn = x/||x|| -> bf16 plane
__global__ void normalize_kernel(const float* __restrict__ x,
                                 uint16_t* __restrict__ xnbf) {
  __shared__ float red[256];
  const int row = blockIdx.x;
  const int t = threadIdx.x;
  const float* xr = x + (size_t)row * DIM;
  float a = xr[t], b = xr[t + 256];
  red[t] = a * a + b * b;
  __syncthreads();
  for (int s = 128; s > 0; s >>= 1) {
    if (t < s) red[t] += red[t + s];
    __syncthreads();
  }
  float rinv = rsqrtf(red[0]);
  uint16_t* o = xnbf + (size_t)row * DIM;
  o[t]       = (uint16_t)f2bf(a * rinv);
  o[t + 256] = (uint16_t)f2bf(b * rinv);
}

// -------------------------------------------------------- W f32 -> bf16 plane
__global__ void convert_w_kernel(const float* __restrict__ W,
                                 uint16_t* __restrict__ wbf) {
  size_t i = (size_t)blockIdx.x * 256 + threadIdx.x;      // one per 8 floats
  const U128* p = (const U128*)(W + i * 8);
  U128 f0 = p[0], f1 = p[1];
  U128 o;
  o.x = bf_rne(f0.x) | (bf_rne(f0.y) << 16);
  o.y = bf_rne(f0.z) | (bf_rne(f0.w) << 16);
  o.z = bf_rne(f1.x) | (bf_rne(f1.y) << 16);
  o.w = bf_rne(f1.z) | (bf_rne(f1.w) << 16);
  *(U128*)(wbf + i * 8) = o;
}

// --------------------------------- target[i] = dot(xn_bf16[i], W_bf16[y_i])
__global__ void target_kernel(const uint16_t* __restrict__ xnbf,
                              const uint16_t* __restrict__ wbf,
                              const int* __restrict__ labels,
                              float* __restrict__ target) {
  const int lane = threadIdx.x & 31;
  const int row  = blockIdx.x * 8 + (threadIdx.x >> 5);   // one wave per row
  const uint16_t* xr = xnbf + (size_t)row * DIM + lane * 16;
  const uint16_t* wr = wbf + (size_t)labels[row] * DIM + lane * 16;
  U128 xa = *(const U128*)(xr), xb = *(const U128*)(xr + 8);
  U128 wa = *(const U128*)(wr), wb = *(const U128*)(wr + 8);
  const uint32_t* px = &xa.x;
  const uint32_t* pw = &wa.x;
  float s = 0.0f;
#pragma unroll
  for (int j = 0; j < 8; ++j) {   // xa..xb / wa..wb are contiguous members
    uint32_t ux = (j < 4) ? px[j] : (&xb.x)[j - 4];
    uint32_t uw = (j < 4) ? pw[j] : (&wb.x)[j - 4];
    s = fmaf(bflo(ux), bflo(uw), s);
    s = fmaf(bfhi(ux), bfhi(uw), s);
  }
  s += __shfl_xor(s, 1, 32);
  s += __shfl_xor(s, 2, 32);
  s += __shfl_xor(s, 4, 32);
  s += __shfl_xor(s, 8, 32);
  s += __shfl_xor(s, 16, 32);
  if (lane == 0) target[row] = s;
}

// ------------------- fused bf16 WMMA GEMM + exp epilogue + per-row atomic sum
// workgroup tile: 128 M x 160 N (8 waves as 4Mx2N; wave = 32M x 80N)
__global__ __launch_bounds__(256) void gemm_exp_kernel(
    const uint16_t* __restrict__ xnbf, const uint16_t* __restrict__ wbf,
    float* __restrict__ rowsum) {
  const int lane = threadIdx.x & 31;
  const int wave = threadIdx.x >> 5;
  const int wm   = wave >> 1;          // 0..3
  const int wn   = wave & 1;           // 0..1
  const int l16  = lane & 15;
  const int half = lane >> 4;

  const int mbase = blockIdx.y * 128 + wm * 32;
  const int cbase = blockIdx.x * 160 + wn * 80;

  // A: 16-bit 16x32 layout -> lane holds row (mbase+l16), 2x16B at K offsets
  // half*8 and half*8+16.  B: lane holds column/W-row (cbase+l16), 2x16B at
  // K offsets half*8 and half*8+16.
  const uint16_t* aptr0 = xnbf + (size_t)(mbase + l16) * DIM + half * 8;
  const uint16_t* aptr1 = aptr0 + (size_t)16 * DIM;
  const uint16_t* bptr  = wbf + (size_t)(cbase + l16) * DIM + half * 8;

  f32x8 zero = {0.f, 0.f, 0.f, 0.f, 0.f, 0.f, 0.f, 0.f};
  f32x8 acc[2][5];
#pragma unroll
  for (int a = 0; a < 2; ++a)
#pragma unroll
    for (int n = 0; n < 5; ++n) acc[a][n] = zero;

  for (int kk = 0; kk < DIM; kk += 32) {
    FragU a0, a1;
    a0.u[0] = *(const U128*)(aptr0 + kk);
    a0.u[1] = *(const U128*)(aptr0 + kk + 16);
    a1.u[0] = *(const U128*)(aptr1 + kk);
    a1.u[1] = *(const U128*)(aptr1 + kk + 16);
#pragma unroll
    for (int n = 0; n < 5; ++n) {
      const uint16_t* bp = bptr + (size_t)n * 16 * DIM + kk;
      FragU b;
      b.u[0] = *(const U128*)(bp);
      b.u[1] = *(const U128*)(bp + 16);
      acc[0][n] = __builtin_amdgcn_wmma_f32_16x16x32_bf16(
          false, a0.v, false, b.v, (short)0, acc[0][n], false, false);
      acc[1][n] = __builtin_amdgcn_wmma_f32_16x16x32_bf16(
          false, a1.v, false, b.v, (short)0, acc[1][n], false, false);
    }
  }

  // epilogue: sum_j exp(s * logit) per row; C/D layout: VGPR r holds row
  // mbase+r (lanes 0-15) and mbase+8+r (lanes 16-31), N = lane&15.
#pragma unroll
  for (int a = 0; a < 2; ++a) {
    f32x8 es = zero;
#pragma unroll
    for (int n = 0; n < 5; ++n)
#pragma unroll
      for (int i = 0; i < 8; ++i) es[i] += __expf(acc[a][n][i] * SSCALE);
    const int mb = mbase + a * 16;
#pragma unroll
    for (int r = 0; r < 8; ++r) {
      float v = es[r];
      v += __shfl_xor(v, 1, 32);
      v += __shfl_xor(v, 2, 32);
      v += __shfl_xor(v, 4, 32);
      v += __shfl_xor(v, 8, 32);     // halves stay independent
      if (lane == 0)       atomicAdd(rowsum + mb + r, v);
      else if (lane == 16) atomicAdd(rowsum + mb + 8 + r, v);
    }
  }
}

// ------------------------------------------------------------- final scalar
__global__ void loss_kernel(const float* __restrict__ rowsum,
                            const float* __restrict__ target,
                            float* __restrict__ out) {
  __shared__ float red[256];
  float local = 0.0f;
  for (int i = threadIdx.x; i < NROWS; i += 256) {
    float t   = target[i];
    float num = SSCALE * (t - MARGINC);
    float excl = rowsum[i] - __expf(SSCALE * t);
    float L = num - logf(__expf(num) + excl);
    local += L;
  }
  red[threadIdx.x] = local;
  __syncthreads();
  for (int s = 128; s > 0; s >>= 1) {
    if (threadIdx.x < s) red[threadIdx.x] += red[threadIdx.x + s];
    __syncthreads();
  }
  if (threadIdx.x == 0) out[0] = -red[0] / (float)NROWS;
}

extern "C" void kernel_launch(void* const* d_in, const int* in_sizes, int n_in,
                              void* d_out, int out_size, void* d_ws,
                              size_t ws_size, hipStream_t stream) {
  (void)in_sizes; (void)n_in; (void)out_size; (void)ws_size;
  const float* x      = (const float*)d_in[0];
  const int*   labels = (const int*)d_in[1];
  const float* W      = (const float*)d_in[2];
  float* out = (float*)d_out;

  // workspace layout: rowsum(16KB) | target(16KB) | xn_bf16(4MB) | W_bf16(102.4MB)
  char* ws = (char*)d_ws;
  float*    rowsum = (float*)ws;
  float*    target = (float*)(ws + (16 << 10));
  uint16_t* xnbf   = (uint16_t*)(ws + (32 << 10));
  uint16_t* wbf    = (uint16_t*)(ws + (32 << 10) + (size_t)NROWS * DIM * 2);

  zero_kernel<<<NROWS / 256, 256, 0, stream>>>(rowsum);
  normalize_kernel<<<NROWS, 256, 0, stream>>>(x, xnbf);
  convert_w_kernel<<<(NCLS * DIM / 8) / 256, 256, 0, stream>>>(W, wbf);
  target_kernel<<<NROWS / 8, 256, 0, stream>>>(xnbf, wbf, labels, target);
  gemm_exp_kernel<<<dim3(NCLS / 160, NROWS / 128), 256, 0, stream>>>(xnbf, wbf,
                                                                     rowsum);
  loss_kernel<<<1, 256, 0, stream>>>(rowsum, target, out);
}
